// MultiscaleShapeletCSC_25555055411355
// MI455X (gfx1250) — matrix operand order repaired and checked by reading, not compiled, not run on hardware
//
#include <hip/hip_runtime.h>

#define T_LEN   16352
#define B_N     16
#define CIN_N   12
#define NA_N    128
#define ATOM_N  64
#define L_LAT   512
#define ZPAD    576          // 31 left pad + 512 + 33 (round up)
#define NREC    (B_N * CIN_N * T_LEN)   // 3,139,584
#define PB      2            // batches per k_conv block (A-fragment reuse)

typedef __attribute__((ext_vector_type(16))) __bf16 v16bf;
typedef __attribute__((ext_vector_type(8)))  float  v8f;

union Frag { v16bf v; unsigned int u[8]; };

__device__ __forceinline__ unsigned short f2bf(float f) {
  unsigned int u = __float_as_uint(f);
  u += 0x7FFFu + ((u >> 16) & 1u);          // round-to-nearest-even
  return (unsigned short)(u >> 16);
}

// ---------------- zero init: tlr (1M floats) + l1 accumulator ----------------
__global__ void k_zero(float* __restrict__ tlr, float* __restrict__ l1) {
  int i = blockIdx.x * blockDim.x + threadIdx.x;
  if (i < B_N * NA_N * L_LAT) tlr[i] = 0.0f;
  if (i == B_N * NA_N * L_LAT) l1[0] = 0.0f;
}

// ---------------- per-(b,c) mean / std over T ----------------
__global__ void k_norm(const float* __restrict__ x, float* __restrict__ ms) {
  __shared__ float s_sum[256], s_sq[256];
  int row = blockIdx.x;                       // 0..191  (b*12 + c)
  const float* xr = x + (size_t)row * T_LEN;
  float s = 0.f, q = 0.f;
  for (int t = threadIdx.x; t < T_LEN; t += 256) { float v = xr[t]; s += v; q += v * v; }
  s_sum[threadIdx.x] = s; s_sq[threadIdx.x] = q;
  __syncthreads();
  for (int off = 128; off > 0; off >>= 1) {
    if (threadIdx.x < off) { s_sum[threadIdx.x] += s_sum[threadIdx.x + off];
                             s_sq[threadIdx.x]  += s_sq[threadIdx.x + off]; }
    __syncthreads();
  }
  if (threadIdx.x == 0) {
    float mean = s_sum[0] / (float)T_LEN;
    float var  = s_sq[0] / (float)T_LEN - mean * mean;
    ms[row * 2 + 0] = mean;
    ms[row * 2 + 1] = sqrtf(var + 1e-5f);
  }
}

// ---------------- stem conv1 (stride 32, pad 32/32) + ReLU ----------------
__global__ void k_stem1(const float* __restrict__ x, const float* __restrict__ ms,
                        const float* __restrict__ w1, const float* __restrict__ b1,
                        float* __restrict__ h) {
  int i = blockIdx.x * blockDim.x + threadIdx.x;     // b*64*512 + co*512 + l
  int l  = i & (L_LAT - 1);
  int co = (i >> 9) & 63;
  int b  = i >> 15;
  int t0 = l * 32 - 32;
  float acc = b1[co];
  for (int ci = 0; ci < CIN_N; ++ci) {
    float mean = ms[(b * CIN_N + ci) * 2 + 0];
    float rstd = 1.0f / ms[(b * CIN_N + ci) * 2 + 1];
    const float* xr = x  + ((size_t)(b * CIN_N + ci)) * T_LEN;
    const float* wr = w1 + ((size_t)(co * CIN_N + ci)) * ATOM_N;
    for (int k = 0; k < ATOM_N; ++k) {
      int t = t0 + k;
      if (t >= 0 && t < T_LEN) acc += wr[k] * ((xr[t] - mean) * rstd);
    }
  }
  h[i] = acc > 0.f ? acc : 0.f;
}

// ---------------- stem conv2 (1x1, 64->128) * stem_scale ----------------
__global__ void k_stem2(const float* __restrict__ h, const float* __restrict__ w2,
                        const float* __restrict__ b2, const float* __restrict__ sscale,
                        float* __restrict__ zraw) {
  int i = blockIdx.x * blockDim.x + threadIdx.x;     // b*128*512 + a*512 + l
  int l = i & (L_LAT - 1);
  int a = (i >> 9) & (NA_N - 1);
  int b = i >> 16;
  const float* hr = h + ((size_t)b * 64) * L_LAT + l;
  const float* wr = w2 + a * 64;
  float acc = b2[a];
  for (int c = 0; c < 64; ++c) acc += hr[(size_t)c * L_LAT] * wr[c];
  zraw[i] = acc * sscale[0];
}

// ---------------- normalize shapelets, convert to bf16 ----------------
__global__ void k_shpnorm(const float* __restrict__ shp, unsigned short* __restrict__ shpbf) {
  int i = blockIdx.x * blockDim.x + threadIdx.x;     // ao*128 + ai  (16384)
  const float* r = shp + (size_t)i * ATOM_N;
  float q = 0.f;
  for (int k = 0; k < ATOM_N; ++k) q += r[k] * r[k];
  float n = sqrtf(q); n = n < 1e-8f ? 1e-8f : n;
  float rn = 1.0f / n;
  unsigned short* o = shpbf + (size_t)i * ATOM_N;
  for (int k = 0; k < ATOM_N; ++k) o[k] = f2bf(r[k] * rn);
}

// ---------------- decoder weight prefix sums over taps ----------------
__global__ void k_wcum(const float* __restrict__ dw, float* __restrict__ wcum) {
  int i = blockIdx.x * blockDim.x + threadIdx.x;     // c*128 + a (1536)
  const float* r = dw + (size_t)i * ATOM_N;
  float* o = wcum + (size_t)i * (ATOM_N + 1);
  float s = 0.f; o[0] = 0.f;
  for (int k = 0; k < ATOM_N; ++k) { s += r[k]; o[k + 1] = s; }
}

// ------- pool residual by `scale`, upsample to padded bf16 signal, L1 -------
__global__ void k_poolup(const float* __restrict__ zraw, const float* __restrict__ tlr,
                         unsigned short* __restrict__ zup, float* __restrict__ l1,
                         int scale) {
  __shared__ float red[256];
  int i = blockIdx.x * blockDim.x + threadIdx.x;     // b*128*576 + ai*576 + pos
  int pos = i % ZPAD;
  int ai  = (i / ZPAD) & (NA_N - 1);
  int b   = i / (ZPAD * NA_N);
  int l = pos - 31;
  float val = 0.f, contrib = 0.f;
  if (l >= 0 && l < L_LAT) {
    int lbase = (l / scale) * scale;
    size_t base = ((size_t)b * NA_N + ai) * L_LAT + lbase;
    float s = 0.f;
    for (int j = 0; j < scale; ++j) s += zraw[base + j] - tlr[base + j];
    val = s / (float)scale;
    if ((l % scale) == 0)
      contrib = fabsf(val) / (float)(B_N * NA_N * (L_LAT / scale));
  }
  zup[i] = f2bf(val);
  red[threadIdx.x] = contrib;
  __syncthreads();
  for (int off = 128; off > 0; off >>= 1) {
    if (threadIdx.x < off) red[threadIdx.x] += red[threadIdx.x + off];
    __syncthreads();
  }
  if (threadIdx.x == 0 && red[0] != 0.f) atomicAdd(l1, red[0]);
}

// ---------------- WMMA shapelet conv: tlr += conv(zup, shp) ----------------
// block = 256 thr (8 waves) -> (batch-pair, 16-col tile); wave w -> ao rows 16w..16w+15.
// A fragment loaded once per K-chunk, fed to 2 WMMAs (one per batch's LDS window).
__global__ void __launch_bounds__(256)
k_conv(const unsigned short* __restrict__ shpbf, const unsigned short* __restrict__ zup,
       float* __restrict__ tlr) {
  __shared__ unsigned short seg[PB][NA_N * 80];      // 2 sliding windows, 40 KB
  int bp = blockIdx.x >> 5;                          // batch pair 0..7
  int lt = (blockIdx.x & 31) << 4;
  int tid  = threadIdx.x;
  int w    = tid >> 5;
  int lane = tid & 31;
  int hl   = lane >> 4;                              // half-wave select
  int col  = lane & 15;

  // stage the 128 x 80 bf16 window [lt, lt+79] for both batches into LDS
  for (int idx = tid; idx < PB * NA_N * 80; idx += 256) {
    int pb  = idx / (NA_N * 80);
    int r   = idx - pb * (NA_N * 80);
    int ai  = r / 80, off = r - ai * 80;
    seg[pb][ai * 80 + off] =
        zup[(((size_t)(bp * PB + pb)) * NA_N + ai) * ZPAD + lt + off];
  }
  __syncthreads();

  const unsigned short* Arow = shpbf + ((size_t)(16 * w + col)) * (NA_N * ATOM_N);
  v8f acc0 = {}, acc1 = {};
  for (int c = 0; c < 256; ++c) {                    // K = 8192 in chunks of 32
    int kc = c << 5;
    Frag a, b0, b1;
#pragma unroll
    for (int v = 0; v < 8; ++v) {                    // A: 16x32 bf16 layout
      int k0 = kc + ((v >> 2) << 4) + (hl << 3) + ((v & 3) << 1);
      a.u[v] = *(const unsigned int*)(Arow + k0);    // packed K pair, dword load
    }
    int ai = c >> 1, tap = (c & 1) << 5;
    int pbase = tap + col + (hl << 3);
    const unsigned short* s0 = &seg[0][ai * 80];
    const unsigned short* s1 = &seg[1][ai * 80];
#pragma unroll
    for (int v = 0; v < 8; ++v) {                    // B: 32x16 bf16 layout
      int p = pbase + ((v >> 2) << 4) + ((v & 3) << 1); // K pair, same order as A
      b0.u[v] = (unsigned int)s0[p] | ((unsigned int)s0[p + 1] << 16);
      b1.u[v] = (unsigned int)s1[p] | ((unsigned int)s1[p + 1] << 16);
    }
    acc0 = __builtin_amdgcn_wmma_f32_16x16x32_bf16(
        false, a.v, false, b0.v, (short)0, acc0, false, false);
    acc1 = __builtin_amdgcn_wmma_f32_16x16x32_bf16(
        false, a.v, false, b1.v, (short)0, acc1, false, false);
  }
#pragma unroll
  for (int v = 0; v < 8; ++v) {                      // D: M = hl*8+v, N = col
    int rowOut = 16 * w + hl * 8 + v;
    size_t o0 = (((size_t)(bp * PB + 0)) * NA_N + rowOut) * L_LAT + lt + col;
    size_t o1 = (((size_t)(bp * PB + 1)) * NA_N + rowOut) * L_LAT + lt + col;
    tlr[o0] += acc0[v];
    tlr[o1] += acc1[v];
  }
}

// ------- decoder: folded conv over piecewise-constant upsample + denorm -------
__global__ void k_dec(const float* __restrict__ tlr, const float* __restrict__ wcum,
                      const float* __restrict__ db, const float* __restrict__ ms,
                      float* __restrict__ out) {
  int i = blockIdx.x * blockDim.x + threadIdx.x;     // b*12*T + c*T + t
  int t = i % T_LEN;
  int c = (i / T_LEN) % CIN_N;
  int b = i / (T_LEN * CIN_N);
  float mean = ms[(b * CIN_N + c) * 2 + 0];
  float sd   = ms[(b * CIN_N + c) * 2 + 1];
  int uStart = t - 31;
  int fLo = uStart < 0 ? 0 : (uStart >> 5);
  int fHi = (t + 32) >> 5;                           // <= 511
  float acc = db[c];
  for (int a = 0; a < NA_N; ++a) {
    const float* wc = wcum + ((size_t)c * NA_N + a) * (ATOM_N + 1);
    const float* tl = tlr  + ((size_t)b * NA_N + a) * L_LAT;
    float s = 0.f;
    for (int f = fLo; f <= fHi; ++f) {
      int kLo = (f << 5) - uStart;      if (kLo < 0)  kLo = 0;
      int kHi = (f << 5) + 31 - uStart; if (kHi > 63) kHi = 63;
      s += tl[f] * (wc[kHi + 1] - wc[kLo]);
    }
    acc += s;
  }
  out[i] = acc * sd + mean;
}

__global__ void k_final(const float* __restrict__ l1, float* __restrict__ out) {
  out[NREC]     = 0.0f;                 // total_vq_loss
  out[NREC + 1] = 0.01f * l1[0];        // total_l1 * SPARSITY_WEIGHT
}

extern "C" void kernel_launch(void* const* d_in, const int* in_sizes, int n_in,
                              void* d_out, int out_size, void* d_ws, size_t ws_size,
                              hipStream_t stream) {
  const float* x      = (const float*)d_in[0];
  const float* w1     = (const float*)d_in[1];
  const float* b1     = (const float*)d_in[2];
  const float* w2     = (const float*)d_in[3];
  const float* b2     = (const float*)d_in[4];
  const float* sscale = (const float*)d_in[5];
  const float* shp    = (const float*)d_in[6];
  const float* dw     = (const float*)d_in[7];
  const float* db     = (const float*)d_in[8];
  float* out = (float*)d_out;
  char* ws = (char*)d_ws;

  size_t o = 0;
  float*          ms    = (float*)(ws + o);          o += 2048;                       // 192*2 f32
  float*          h     = (float*)(ws + o);          o += (size_t)B_N*64*L_LAT*4;     // 2 MB
  float*          zraw  = (float*)(ws + o);          o += (size_t)B_N*NA_N*L_LAT*4;   // 4 MB
  float*          tlr   = (float*)(ws + o);          o += (size_t)B_N*NA_N*L_LAT*4;   // 4 MB
  unsigned short* shpbf = (unsigned short*)(ws + o); o += (size_t)NA_N*NA_N*ATOM_N*2; // 2 MB
  unsigned short* zup   = (unsigned short*)(ws + o); o += (size_t)B_N*NA_N*ZPAD*2;    // 2.25 MB
  float*          wcum  = (float*)(ws + o);          o += (size_t)CIN_N*NA_N*(ATOM_N+1)*4;
  float*          l1    = (float*)(ws + o);          o += 256;

  k_zero   <<<(B_N*NA_N*L_LAT + 256) / 256, 256, 0, stream>>>(tlr, l1);
  k_norm   <<<B_N * CIN_N, 256, 0, stream>>>(x, ms);
  k_stem1  <<<(B_N*64*L_LAT) / 256, 256, 0, stream>>>(x, ms, w1, b1, h);
  k_stem2  <<<(B_N*NA_N*L_LAT) / 256, 256, 0, stream>>>(h, w2, b2, sscale, zraw);
  k_shpnorm<<<(NA_N*NA_N) / 256, 256, 0, stream>>>(shp, shpbf);
  k_wcum   <<<(CIN_N*NA_N) / 256, 256, 0, stream>>>(dw, wcum);

  const int scales[3] = {4, 2, 1};
  for (int i = 0; i < 3; ++i) {
    k_poolup<<<(B_N*NA_N*ZPAD) / 256, 256, 0, stream>>>(zraw, tlr, zup, l1, scales[i]);
    k_conv  <<<(B_N / PB) * (L_LAT / 16), 256, 0, stream>>>(shpbf, zup, tlr);
  }

  k_dec  <<<NREC / 256, 256, 0, stream>>>(tlr, wcum, db, ms, out);
  k_final<<<1, 1, 0, stream>>>(l1, out);
}
